// HashLayerFFN_78847009620271
// MI455X (gfx1250) — compile-verified, matrix operand order checked by reading, not run on hardware
//
#include <hip/hip_runtime.h>
#include <hip/hip_bf16.h>
#include <stdint.h>

// Problem constants (B=2,S=1024 -> T=2048 tokens), D=1024, H=2048, E=8 experts.
#define T_TOK 2048
#define DD    1024
#define HH    2048
#define EE    8

typedef __attribute__((ext_vector_type(16))) __bf16 v16bf;
typedef __attribute__((ext_vector_type(8)))  __bf16 v8bf;
typedef __attribute__((ext_vector_type(8)))  float  v8f;
typedef __attribute__((ext_vector_type(4)))  float  v4f;
typedef __attribute__((ext_vector_type(4)))  unsigned int v4u;

union FragBF { v16bf v; v8bf h[2]; };
union PackBF { __bf16 b[4]; uint64_t u; };

// ---------------- Routing: meta layout (ints): ----------------
//   meta[0..7]   = counts per expert
//   meta[8..15]  = cursors (scratch for compaction)
//   meta[16..23] = exclusive prefix offsets
//   meta[32..32+T_TOK) = compacted token permutation (grouped by expert)

__global__ void route_init_kernel(int* meta) {
    if (threadIdx.x < 32) meta[threadIdx.x] = 0;
}

__global__ void route_count_kernel(const int* __restrict__ orig,
                                   const int* __restrict__ hashm,
                                   int* __restrict__ meta) {
    int t = blockIdx.x * 256 + threadIdx.x;
    if (t < T_TOK) {
        int e = hashm[orig[t]] & (EE - 1);
        atomicAdd(&meta[e], 1);
    }
}

__global__ void route_scan_kernel(int* meta) {
    if (threadIdx.x == 0) {
        int s = 0;
        for (int i = 0; i < EE; ++i) { meta[16 + i] = s; s += meta[i]; }
    }
}

__global__ void route_assign_kernel(const int* __restrict__ orig,
                                    const int* __restrict__ hashm,
                                    int* __restrict__ meta) {
    int t = blockIdx.x * 256 + threadIdx.x;
    if (t < T_TOK) {
        int e = hashm[orig[t]] & (EE - 1);
        int p = atomicAdd(&meta[8 + e], 1);
        meta[32 + meta[16 + e] + p] = t;
    }
}

// ---------------- Layer 1: hidden = relu(X * W1^T + b1), bf16 out ----------------
// Block: 256 threads = 8 waves in 4(M)x2(N); block tile 64(M) x 64(N); K staged
// 64-wide in LDS; each wave: 1 A frag reused over 2 B frags -> 4 WMMA / stage.
__global__ __launch_bounds__(256) void ffn1_kernel(const float* __restrict__ x,
                                                   const float* __restrict__ W1,
                                                   const float* __restrict__ b1,
                                                   const int* __restrict__ meta,
                                                   __bf16* __restrict__ hidden) {
    const int e   = blockIdx.z;
    const int cnt = meta[e];
    const int off = meta[16 + e];
    const int m0  = blockIdx.y * 64;
    if (m0 >= cnt) return;
    const int n0  = blockIdx.x * 64;

    __shared__ __bf16 lx[64][64];
    __shared__ __bf16 lw[64][64];
    __shared__ int    tok[64];

    const int tid = threadIdx.x;
    if (tid < 64) {
        int r = m0 + tid;
        if (r > cnt - 1) r = cnt - 1;      // clamp: rows are independent, tail
        tok[tid] = meta[32 + off + r];     // rows compute garbage, never stored
    }
    __syncthreads();

    const int lane = tid & 31;
    const int wave = tid >> 5;
    const int mi   = wave >> 1;      // 0..3  (M sub-tile)
    const int ni   = wave & 1;       // 0..1  (N sub-tile pair)
    const int half = lane >> 4;      // K-half select per ISA 16-bit A layout
    const int l15  = lane & 15;

    // Staging map: 16 rows x 16 segments(4 floats) per pass, 4 passes.
    const int srow = tid >> 4;          // 0..15
    const int sseg = (tid & 15) * 4;    // 0..60

    const float* xrow[4];
    #pragma unroll
    for (int p = 0; p < 4; ++p)
        xrow[p] = x + (size_t)tok[srow + p * 16] * DD;

    const float* wbase = W1 + (size_t)e * HH * DD + (size_t)n0 * DD;

    v8f acc[2] = {};

    for (int k0 = 0; k0 < DD; k0 += 64) {
        #pragma unroll
        for (int p = 0; p < 4; ++p) {
            int r = srow + p * 16;
            v4f v = *(const v4f*)(xrow[p] + k0 + sseg);
            PackBF pa;
            pa.b[0] = (__bf16)v[0]; pa.b[1] = (__bf16)v[1];
            pa.b[2] = (__bf16)v[2]; pa.b[3] = (__bf16)v[3];
            *(uint64_t*)&lx[r][sseg] = pa.u;
            v4f w = *(const v4f*)(wbase + (size_t)r * DD + k0 + sseg);
            PackBF pw;
            pw.b[0] = (__bf16)w[0]; pw.b[1] = (__bf16)w[1];
            pw.b[2] = (__bf16)w[2]; pw.b[3] = (__bf16)w[3];
            *(uint64_t*)&lw[r][sseg] = pw.u;
        }
        __syncthreads();

        #pragma unroll
        for (int ks = 0; ks < 64; ks += 32) {
            FragBF a;
            a.h[0] = *(const v8bf*)&lx[mi * 16 + l15][ks + half * 8];
            a.h[1] = *(const v8bf*)&lx[mi * 16 + l15][ks + 16 + half * 8];
            #pragma unroll
            for (int nj = 0; nj < 2; ++nj) {
                FragBF b;
                b.h[0] = *(const v8bf*)&lw[ni * 32 + nj * 16 + l15][ks + half * 8];
                b.h[1] = *(const v8bf*)&lw[ni * 32 + nj * 16 + l15][ks + 16 + half * 8];
                acc[nj] = __builtin_amdgcn_wmma_f32_16x16x32_bf16(
                    false, a.v, false, b.v, (short)0, acc[nj], false, false);
            }
        }
        __syncthreads();
    }

    // Epilogue: bias + relu, store hidden (bf16) by compact slot.
    // Fast path: full tile (no per-row bounds checks / exec masking).
    const int mbase = m0 + mi * 16 + half * 8;
    if (m0 + 64 <= cnt) {
        #pragma unroll
        for (int nj = 0; nj < 2; ++nj) {
            const int   nc   = n0 + ni * 32 + nj * 16 + l15;
            const float bias = b1[e * HH + nc];
            __bf16* hp = hidden + (size_t)(off + mbase) * HH + nc;
            #pragma unroll
            for (int i = 0; i < 8; ++i) {
                float v = acc[nj][i] + bias;
                v = v > 0.f ? v : 0.f;
                hp[(size_t)i * HH] = (__bf16)v;
            }
        }
    } else {
        #pragma unroll
        for (int nj = 0; nj < 2; ++nj) {
            const int   nc   = n0 + ni * 32 + nj * 16 + l15;
            const float bias = b1[e * HH + nc];
            #pragma unroll
            for (int i = 0; i < 8; ++i) {
                int row = mbase + i;
                if (row < cnt) {
                    float v = acc[nj][i] + bias;
                    v = v > 0.f ? v : 0.f;
                    hidden[(size_t)(off + row) * HH + nc] = (__bf16)v;
                }
            }
        }
    }
}

// ---------------- Layer 2: out = hidden * W2^T + b2, scatter by token ----------------
__global__ __launch_bounds__(256) void ffn2_kernel(const __bf16* __restrict__ hidden,
                                                   const float* __restrict__ W2,
                                                   const float* __restrict__ b2,
                                                   const int* __restrict__ meta,
                                                   float* __restrict__ out) {
    const int e   = blockIdx.z;
    const int cnt = meta[e];
    const int off = meta[16 + e];
    const int m0  = blockIdx.y * 64;
    if (m0 >= cnt) return;
    const int n0  = blockIdx.x * 64;

    __shared__ __bf16 lh[64][64];
    __shared__ __bf16 lw[64][64];
    __shared__ int    tok[64];

    const int tid = threadIdx.x;
    if (tid < 64) {
        int r = m0 + tid;
        if (r > cnt - 1) r = cnt - 1;
        tok[tid] = meta[32 + off + r];
    }
    __syncthreads();

    const int lane = tid & 31;
    const int wave = tid >> 5;
    const int mi   = wave >> 1;
    const int ni   = wave & 1;
    const int half = lane >> 4;
    const int l15  = lane & 15;

    // A staging: bf16 straight copy, 32 rows x 8 segs(16B) per pass, 2 passes.
    const int crow = tid >> 3;          // 0..31
    const int cseg = (tid & 7) * 8;     // bf16 element offset, 16B chunks
    const __bf16* hrowp[2];
    #pragma unroll
    for (int p = 0; p < 2; ++p) {
        int r = off + m0 + crow + p * 32;
        if (r > T_TOK - 1) r = T_TOK - 1;   // clamp tail reads
        hrowp[p] = hidden + (size_t)r * HH;
    }

    // W staging map (fp32 -> bf16): 16 rows x 16 segs per pass, 4 passes.
    const int srow = tid >> 4;
    const int sseg = (tid & 15) * 4;
    const float* wbase = W2 + (size_t)e * DD * HH + (size_t)n0 * HH;

    v8f acc[2] = {};

    for (int k0 = 0; k0 < HH; k0 += 64) {
        #pragma unroll
        for (int p = 0; p < 2; ++p)
            *(v4u*)&lh[crow + p * 32][cseg] = *(const v4u*)(hrowp[p] + k0 + cseg);
        #pragma unroll
        for (int p = 0; p < 4; ++p) {
            int r = srow + p * 16;
            v4f w = *(const v4f*)(wbase + (size_t)r * HH + k0 + sseg);
            PackBF pw;
            pw.b[0] = (__bf16)w[0]; pw.b[1] = (__bf16)w[1];
            pw.b[2] = (__bf16)w[2]; pw.b[3] = (__bf16)w[3];
            *(uint64_t*)&lw[r][sseg] = pw.u;
        }
        __syncthreads();

        #pragma unroll
        for (int ks = 0; ks < 64; ks += 32) {
            FragBF a;
            a.h[0] = *(const v8bf*)&lh[mi * 16 + l15][ks + half * 8];
            a.h[1] = *(const v8bf*)&lh[mi * 16 + l15][ks + 16 + half * 8];
            #pragma unroll
            for (int nj = 0; nj < 2; ++nj) {
                FragBF b;
                b.h[0] = *(const v8bf*)&lw[ni * 32 + nj * 16 + l15][ks + half * 8];
                b.h[1] = *(const v8bf*)&lw[ni * 32 + nj * 16 + l15][ks + 16 + half * 8];
                acc[nj] = __builtin_amdgcn_wmma_f32_16x16x32_bf16(
                    false, a.v, false, b.v, (short)0, acc[nj], false, false);
            }
        }
        __syncthreads();
    }

    const int mloc = mi * 16 + half * 8;   // local row base within block tile
    if (m0 + 64 <= cnt) {
        #pragma unroll
        for (int nj = 0; nj < 2; ++nj) {
            const int   nc   = n0 + ni * 32 + nj * 16 + l15;
            const float bias = b2[e * DD + nc];
            #pragma unroll
            for (int i = 0; i < 8; ++i) {
                int t = tok[mloc + i];
                out[(size_t)t * DD + nc] = acc[nj][i] + bias;
            }
        }
    } else {
        #pragma unroll
        for (int nj = 0; nj < 2; ++nj) {
            const int   nc   = n0 + ni * 32 + nj * 16 + l15;
            const float bias = b2[e * DD + nc];
            #pragma unroll
            for (int i = 0; i < 8; ++i) {
                int row = m0 + mloc + i;
                if (row < cnt) {
                    int t = tok[mloc + i];
                    out[(size_t)t * DD + nc] = acc[nj][i] + bias;
                }
            }
        }
    }
}

extern "C" void kernel_launch(void* const* d_in, const int* in_sizes, int n_in,
                              void* d_out, int out_size, void* d_ws, size_t ws_size,
                              hipStream_t stream) {
    (void)in_sizes; (void)n_in; (void)out_size; (void)ws_size;

    const float* x     = (const float*)d_in[0];
    const int*   orig  = (const int*)d_in[1];
    const int*   hashm = (const int*)d_in[2];
    const float* W1    = (const float*)d_in[3];
    const float* b1    = (const float*)d_in[4];
    const float* W2    = (const float*)d_in[5];
    const float* b2    = (const float*)d_in[6];
    float*       out   = (float*)d_out;

    int*    meta   = (int*)d_ws;                              // 32 ints + perm[T]
    __bf16* hidden = (__bf16*)((char*)d_ws + 16384);          // T*H bf16 = 8 MB

    // Routing / compaction
    route_init_kernel<<<1, 32, 0, stream>>>(meta);
    route_count_kernel<<<T_TOK / 256, 256, 0, stream>>>(orig, hashm, meta);
    route_scan_kernel<<<1, 1, 0, stream>>>(meta);
    route_assign_kernel<<<T_TOK / 256, 256, 0, stream>>>(orig, hashm, meta);

    // Layer 1: worst-case M tiles per expert, empty tiles early-exit.
    dim3 g1(HH / 64, T_TOK / 64, EE);   // (32, 32, 8)
    ffn1_kernel<<<g1, 256, 0, stream>>>(x, W1, b1, meta, hidden);

    // Layer 2
    dim3 g2(DD / 64, T_TOK / 64, EE);   // (16, 32, 8)
    ffn2_kernel<<<g2, 256, 0, stream>>>(hidden, W2, b2, meta, out);
}
